// KMeansPalettizedLinear_51264729645536
// MI455X (gfx1250) — matrix operand order repaired and checked by reading, not compiled, not run on hardware
//
#include <hip/hip_runtime.h>

typedef __attribute__((ext_vector_type(4)))  float  v4f;
typedef __attribute__((ext_vector_type(8)))  float  v8f;
typedef __attribute__((ext_vector_type(4)))  int    v4i;
typedef __attribute__((ext_vector_type(2)))  unsigned int v2u;
typedef __attribute__((ext_vector_type(8)))  __bf16 v8bf;
typedef __attribute__((ext_vector_type(16))) __bf16 v16bf;

#define BM 128
#define BN 128
#define BK 32
#define LSTR 40   // ushorts per row: 32 data + 8 pad = 80 B = 20 dwords (5 coprime 16 -> bank-optimal)
#define KDIM 4096
#define NDIM 4096

static __device__ __forceinline__ unsigned short bfbits(__bf16 b) {
  union { __bf16 b; unsigned short u; } cv; cv.b = b; return cv.u;
}
// split fp32 into bf16 hi + bf16 residual, return {hi_bits, lo_bits}
static __device__ __forceinline__ void split_bf16(float x, unsigned short& h, unsigned short& l) {
  __bf16 hb = (__bf16)x;
  h = bfbits(hb);
  l = bfbits((__bf16)(x - (float)hb));
}
static __device__ __forceinline__ v16bf frag_cat(v8bf a, v8bf b) {
  return __builtin_shufflevector(a, b, 0,1,2,3,4,5,6,7,8,9,10,11,12,13,14,15);
}

__launch_bounds__(256)
__global__ void palettized_gemm_bf16x3(const float* __restrict__ X,
                                       const float* __restrict__ lut,
                                       const int*   __restrict__ widx,
                                       const float* __restrict__ bias,
                                       float*       __restrict__ out) {
  __shared__ __align__(16) unsigned short XsH[BM * LSTR];  // 10240 B each
  __shared__ __align__(16) unsigned short XsL[BM * LSTR];
  __shared__ __align__(16) unsigned short WsH[BN * LSTR];  // W kept [N][K]: matches B-frag layout
  __shared__ __align__(16) unsigned short WsL[BN * LSTR];
  __shared__ unsigned int Lp[256];                         // packed {lo16, hi16} palette

  const int tid   = threadIdx.x;
  const int lane  = tid & 31;
  const int wave  = tid >> 5;
  const int waveM = wave >> 2;           // 0..1 -> 64 rows
  const int waveN = wave & 3;            // 0..3 -> 32 cols
  const int l16   = lane & 15;
  const int hiH   = lane >> 4;           // 0 / 1
  const int kA    = hiH ? 8 : 0;         // A-frag: chunks at kA, kA+16 (ushort offsets)
  const int kB    = hiH ? 8 : 0;         // B-frag: chunks at kB, kB+16

  const int bm = blockIdx.y * BM;
  const int bn = blockIdx.x * BN;

  // --- split palette once: Lp[i] = lo_bits<<16 | hi_bits ---
  {
    float v = lut[tid & 255];
    unsigned short h, l; split_bf16(v, h, l);
    Lp[tid & 255] = (unsigned int)h | ((unsigned int)l << 16);
  }

  v8f acc[4][2];
#pragma unroll
  for (int mt = 0; mt < 4; ++mt)
#pragma unroll
    for (int nt = 0; nt < 2; ++nt)
      acc[mt][nt] = (v8f){0.f,0.f,0.f,0.f,0.f,0.f,0.f,0.f};

  const int loadRow = tid >> 3;          // 0..31
  const int loadCol = (tid & 7) << 2;    // 0,4,...,28

  for (int kt = 0; kt < KDIM; kt += BK) {
    if (kt + BK < KDIM) {               // hint next tiles toward GL2/L0
      __builtin_prefetch(&X[(size_t)(bm + loadRow) * KDIM + kt + BK + loadCol], 0, 0);
      __builtin_prefetch(&widx[(size_t)(bn + loadRow) * KDIM + kt + BK + loadCol], 0, 0);
    }
    __syncthreads();   // prior iteration's LDS reads done (covers Lp on iter 0)

    // ---- stage X tile: 128x32 fp32 -> bf16 hi/lo ----
#pragma unroll
    for (int p = 0; p < 4; ++p) {
      const int r = loadRow + p * 32;
      v4f x = *(const v4f*)&X[(size_t)(bm + r) * KDIM + kt + loadCol];
      unsigned short h0,h1,h2,h3,l0,l1,l2,l3;
      split_bf16(x.x,h0,l0); split_bf16(x.y,h1,l1);
      split_bf16(x.z,h2,l2); split_bf16(x.w,h3,l3);
      *(v2u*)&XsH[r * LSTR + loadCol] =
          (v2u){(unsigned)h0 | ((unsigned)h1 << 16), (unsigned)h2 | ((unsigned)h3 << 16)};
      *(v2u*)&XsL[r * LSTR + loadCol] =
          (v2u){(unsigned)l0 | ((unsigned)l1 << 16), (unsigned)l2 | ((unsigned)l3 << 16)};
    }
    // ---- stage + dequant W tile: LDS gather of packed palette ----
#pragma unroll
    for (int p = 0; p < 4; ++p) {
      const int n = loadRow + p * 32;
      v4i i4 = *(const v4i*)&widx[(size_t)(bn + n) * KDIM + kt + loadCol];
      unsigned int p0 = Lp[i4.x], p1 = Lp[i4.y], p2 = Lp[i4.z], p3 = Lp[i4.w];
      *(v2u*)&WsH[n * LSTR + loadCol] =
          (v2u){(p0 & 0xffffu) | (p1 << 16), (p2 & 0xffffu) | (p3 << 16)};
      *(v2u*)&WsL[n * LSTR + loadCol] =
          (v2u){(p0 >> 16) | (p1 & 0xffff0000u), (p2 >> 16) | (p3 & 0xffff0000u)};
    }
    __syncthreads();

    // ---- B fragments (K=32 in one shot), 2 n-tiles, hi+lo ----
    v16bf bh[2], bl[2];
#pragma unroll
    for (int nt = 0; nt < 2; ++nt) {
      const int n = waveN * 32 + nt * 16 + l16;
      const unsigned short* pH = &WsH[n * LSTR + kB];
      const unsigned short* pL = &WsL[n * LSTR + kB];
      bh[nt] = frag_cat(*(const v8bf*)pH, *(const v8bf*)(pH + 16));
      bl[nt] = frag_cat(*(const v8bf*)pL, *(const v8bf*)(pL + 16));
    }
    // ---- A fragments per m-tile, 3-term bf16x3 accumulation ----
#pragma unroll
    for (int mt = 0; mt < 4; ++mt) {
      const int row = waveM * 64 + mt * 16 + l16;
      const unsigned short* pH = &XsH[row * LSTR + kA];
      const unsigned short* pL = &XsL[row * LSTR + kA];
      v16bf ah = frag_cat(*(const v8bf*)pH, *(const v8bf*)(pH + 16));
      v16bf al = frag_cat(*(const v8bf*)pL, *(const v8bf*)(pL + 16));
#pragma unroll
      for (int nt = 0; nt < 2; ++nt) {
        acc[mt][nt] = __builtin_amdgcn_wmma_f32_16x16x32_bf16(
            false, ah, false, bh[nt], (short)0, acc[mt][nt], false, false);
        acc[mt][nt] = __builtin_amdgcn_wmma_f32_16x16x32_bf16(
            false, ah, false, bl[nt], (short)0, acc[mt][nt], false, false);
        acc[mt][nt] = __builtin_amdgcn_wmma_f32_16x16x32_bf16(
            false, al, false, bh[nt], (short)0, acc[mt][nt], false, false);
      }
    }
  }

  // ---- epilogue: bias + store (C/D layout: VGPR r -> M=r (+8 upper half), N=lane%16) ----
  const int hiOff = hiH << 3;
#pragma unroll
  for (int nt = 0; nt < 2; ++nt) {
    const int col = bn + waveN * 32 + nt * 16 + l16;
    const float bv = bias[col];
#pragma unroll
    for (int mt = 0; mt < 4; ++mt) {
      const int rbase = bm + waveM * 64 + mt * 16 + hiOff;
#pragma unroll
      for (int r = 0; r < 8; ++r)
        out[(size_t)(rbase + r) * NDIM + col] = acc[mt][nt][r] + bv;
    }
  }
}

extern "C" void kernel_launch(void* const* d_in, const int* in_sizes, int n_in,
                              void* d_out, int out_size, void* d_ws, size_t ws_size,
                              hipStream_t stream) {
  (void)n_in; (void)out_size; (void)d_ws; (void)ws_size;
  const float* X    = (const float*)d_in[0];   // [B,S,D_IN] fp32
  const float* lut  = (const float*)d_in[1];   // [256] fp32
  const int*   widx = (const int*)d_in[2];     // [D_OUT,D_IN] int32
  const float* bias = (const float*)d_in[3];   // [D_OUT] fp32
  float* out = (float*)d_out;                  // [B,S,D_OUT] fp32

  const int M = in_sizes[0] / KDIM;            // B*S = 8192
  dim3 grid(NDIM / BN, M / BM);                // 32 x 64
  palettized_gemm_bf16x3<<<grid, 256, 0, stream>>>(X, lut, widx, bias, out);
}
